// SparseCutAttention_86887188398245
// MI455X (gfx1250) — matrix operand order repaired and checked by reading, not compiled, not run on hardware
//
#include <hip/hip_runtime.h>
#include <hip/hip_bf16.h>
#include <cstdint>
#include <cstddef>

typedef __attribute__((ext_vector_type(16))) _Float16 v16h;
typedef __attribute__((ext_vector_type(8)))  _Float16 v8h;
typedef __attribute__((ext_vector_type(8)))  float    v8f;
typedef __attribute__((ext_vector_type(4)))  unsigned int v4u;
typedef __attribute__((ext_vector_type(8)))  int          v8i;
typedef __attribute__((ext_vector_type(4)))  int          v4i;

#define B_    8
#define C_    512
#define H_    4096
#define HB_   256     // H/BLOCKS
#define NB_   15      // BLOCKS-1
#define CUT_  8
#define M_    2048    // CUT*HB
#define TEMP_ 0.7f
#define NEG_  -1e9f

// per key-chunk (32 keys): 16384 halfs of K-frags + 16384 halfs of V-frags
#define CHUNK_HALFS 32768
#define CHUNK_BYTES 65536

// ---------------- workspace byte offsets (256B aligned) ----------------
#define OFF_SQ    ((size_t)0)                       // 30720 f32
#define OFF_SK    ((size_t)122880)                  // 30720 f32
#define OFF_PERM  ((size_t)245760)                  // 960 f32 (B*15*8)
#define OFF_PART  ((size_t)249600)                  // B*256*2 f32
#define OFF_STAT  ((size_t)265984)                  // B*2 f32
#define OFF_QP    ((size_t)266240)                  // 16,777,216 halfs (32MB)
#define OFF_KVP   (OFF_QP + (size_t)33554432)       // 16,777,216 halfs (32MB)

__device__ inline v16h cat16(v8h lo, v8h hi) {
  return __builtin_shufflevector(lo, hi, 0,1,2,3,4,5,6,7,8,9,10,11,12,13,14,15);
}
__device__ inline v16h ld16(const _Float16* p) {     // global or LDS (generic)
  v8h lo = *(const v8h*)p;
  v8h hi = *(const v8h*)(p + 8);
  return cat16(lo, hi);
}

// ---- Tensor Data Mover: 1-D contiguous copy global -> LDS (2B elements) ----
// D# per CDNA5 ISA ch.8: group0 {count=1 | lds_addr | global_addr | type=2},
// group1 {data_size=2B, tensor_dim0 = tile_dim0 = nelem}, groups 2/3 unused.
// This toolchain exposes the 6-arg builtin: (v4u, v8i, v4i, v4i, v8i, cpol).
__device__ inline void tdm_load_1d(unsigned lds_off, const void* gaddr, unsigned nelem) {
  unsigned long long ga = (unsigned long long)(uintptr_t)gaddr;
  v4u g0;
  g0.x = 1u;                                            // count=1 (user descriptor)
  g0.y = lds_off;                                       // LDS byte address
  g0.z = (unsigned)(ga & 0xFFFFFFFFu);                  // global_addr[31:0]
  g0.w = (unsigned)((ga >> 32) & 0x01FFFFFFu) | (2u << 30);  // [56:32] | type=2
  v8i g1;
  g1[0] = (int)(1u << 16);                              // data_size=1 -> 2 bytes
  g1[1] = (int)((nelem & 0xFFFFu) << 16);               // tensor_dim0[15:0]
  g1[2] = (int)((nelem >> 16) & 0xFFFFu);               // tensor_dim0[31:16]
  g1[3] = (int)((nelem & 0xFFFFu) << 16);               // tile_dim0 = nelem (<=65535)
  g1[4] = 0;                                            // tile_dim1/2 unused
  g1[5] = (int)nelem;                                   // tensor_dim0_stride (unused in 1D)
  g1[6] = 0;
  g1[7] = 0;
  v4i z4 = {0, 0, 0, 0};
  v8i z8 = {0, 0, 0, 0, 0, 0, 0, 0};
  __builtin_amdgcn_tensor_load_to_lds(g0, g1, z4, z4, z8, 0);
}

// ======================================================================
// 1) column means over C for the first 15 blocks of q and k
// ======================================================================
__global__ void mean_kernel(const float* __restrict__ q, const float* __restrict__ k,
                            float* __restrict__ sq, float* __restrict__ sk) {
  int idx = blockIdx.x * blockDim.x + threadIdx.x;   // 0..30719
  int hp  = idx % (NB_ * HB_);
  int b   = idx / (NB_ * HB_);
  const float* qp = q + (size_t)b * C_ * H_ + hp;
  const float* kp = k + (size_t)b * C_ * H_ + hp;
  float aq = 0.f, ak = 0.f;
  for (int c = 0; c < C_; ++c) {
    aq += qp[(size_t)c * H_];
    ak += kp[(size_t)c * H_];
  }
  sq[idx] = aq * (1.0f / C_);
  sk[idx] = ak * (1.0f / C_);
}

// ======================================================================
// 2) R -> gumbel (JAX threefry2x32, key=(0,42)) -> sinkhorn -> perm[:, :, :8]
// ======================================================================
__device__ inline unsigned rotl32(unsigned x, int n) { return (x << n) | (x >> (32 - n)); }

__global__ void sinkhorn_kernel(const float* __restrict__ sq, const float* __restrict__ sk,
                                float* __restrict__ perm8) {
  __shared__ float rS[NB_ * NB_];   // 225
  __shared__ float tmp[NB_];
  const int b = blockIdx.x, t = threadIdx.x;

  if (t < NB_ * NB_) {
    int i = t / NB_, j = t % NB_;
    const float* a = sq + b * (NB_ * HB_) + i * HB_;
    const float* c = sk + b * (NB_ * HB_) + j * HB_;
    float dot = 0.f;
    for (int h = 0; h < HB_; ++h) dot += a[h] * c[h];
    float R = dot * 0.04419417382415922f;                 // 512^-0.5
    float r = (R > 0.f) ? logf(R) : NEG_;

    // --- JAX threefry2x32: counts = iota(1800), split halves ---
    unsigned e  = (unsigned)(b * 225 + t);
    unsigned x0 = (e < 900u) ? e : (e - 900u);
    unsigned x1 = x0 + 900u;
    unsigned ks[3] = {0u, 42u, 0u ^ 42u ^ 0x1BD11BDAu};
    x0 += ks[0]; x1 += ks[1];
    const int rot[8] = {13, 15, 26, 6, 17, 29, 16, 24};
    #pragma unroll
    for (int s = 0; s < 5; ++s) {
      #pragma unroll
      for (int qd = 0; qd < 4; ++qd) {
        x0 += x1; x1 = rotl32(x1, rot[4 * (s & 1) + qd]); x1 ^= x0;
      }
      x0 += ks[(s + 1) % 3];
      x1 += ks[(s + 2) % 3] + (unsigned)(s + 1);
    }
    unsigned bits = (e < 900u) ? x0 : x1;
    float u = __uint_as_float(0x3f800000u | (bits >> 9)) - 1.0f;   // [0,1)
    float g = -logf(-logf(u + 1e-6f) + 1e-6f);
    rS[t] = (r + g) * (1.0f / TEMP_);
  }
  __syncthreads();

  for (int it = 0; it < 8; ++it) {
    if (t < NB_) {           // logsumexp over axis=2 (j), per row t
      float mx = -3.4e38f;
      for (int j = 0; j < NB_; ++j) mx = fmaxf(mx, rS[t * NB_ + j]);
      float s = 0.f;
      for (int j = 0; j < NB_; ++j) s += expf(rS[t * NB_ + j] - mx);
      tmp[t] = mx + logf(s);
    }
    __syncthreads();
    if (t < NB_ * NB_) rS[t] -= tmp[t / NB_];
    __syncthreads();
    if (t < NB_) {           // logsumexp over axis=1 (i), per col t
      float mx = -3.4e38f;
      for (int i = 0; i < NB_; ++i) mx = fmaxf(mx, rS[i * NB_ + t]);
      float s = 0.f;
      for (int i = 0; i < NB_; ++i) s += expf(rS[i * NB_ + t] - mx);
      tmp[t] = mx + logf(s);
    }
    __syncthreads();
    if (t < NB_ * NB_) rS[t] -= tmp[t % NB_];
    __syncthreads();
  }
  if (t < NB_ * CUT_) {      // perm8[b][n][j], j<8
    int n = t / CUT_, j = t % CUT_;
    perm8[b * (NB_ * CUT_) + t] = expf(rS[n * NB_ + j]);
  }
}

// ======================================================================
// 3a) pack Q (f32 [b][c][h]) into f16 WMMA A-fragments
//     QP[b][qt(256)][kt(16)][lane(32)][half(16)]
//     A 16x32 f16 layout: lane = mm + 16*((kk/8)&1); half = (kk&7) + 8*(kk>>4)
// ======================================================================
__global__ void qpack_kernel(const float* __restrict__ q, _Float16* __restrict__ QP) {
  size_t idx = (size_t)blockIdx.x * 256 + threadIdx.x;       // 16,777,216
  int h = (int)(idx % H_);
  int c = (int)((idx / H_) % C_);
  int b = (int)(idx / ((size_t)H_ * C_));
  float v = q[idx];
  int qt = h >> 4, mm = h & 15;
  int kt = c >> 5, kk = c & 31;
  int lane = mm + 16 * ((kk >> 3) & 1);
  int half = (kk & 7) + 8 * (kk >> 4);
  size_t dst = ((((size_t)b * 256 + qt) * 16 + kt) * 32 + lane) * 16 + half;
  QP[dst] = (_Float16)v;
}

// ======================================================================
// 3b) block-mix k/v with perm; pack both into the chunk-contiguous KVP
//     buffer (per 32-key chunk: 16384 halfs K-frags then 16384 halfs V-frags)
//     K (as K^T: B[k=c][n=key]) frag (mtL,kt): ((mtL*16+kt)*32+lane)*16+half
//     V (B[k=key][n=c])        frag nt:  16384 + (nt*32+lane)*16+half
//     B 32x16 f16 layout: lane = nn + 16*((kk/8)&1); half = (kk&7) + 8*(kk>>4)
// ======================================================================
__global__ void kvpack_kernel(const float* __restrict__ x, const float* __restrict__ perm8,
                              _Float16* __restrict__ KVP, int isV) {
  __shared__ float P[NB_][CUT_];
  const int tid = threadIdx.x;
  size_t gid = (size_t)blockIdx.x * 256 + tid;               // 1,048,576
  int h = (int)(gid % HB_);
  int c = (int)((gid / HB_) % C_);
  int b = (int)(gid / ((size_t)HB_ * C_));                   // constant per block
  if (tid < NB_ * CUT_) ((float*)P)[tid] = perm8[b * (NB_ * CUT_) + tid];
  __syncthreads();

  float acc[CUT_];
  #pragma unroll
  for (int j = 0; j < CUT_; ++j) acc[j] = 0.f;
  const float* xp = x + (size_t)b * C_ * H_ + (size_t)c * H_ + h;
  #pragma unroll
  for (int n = 0; n < NB_; ++n) {
    float v = xp[n * HB_];
    #pragma unroll
    for (int j = 0; j < CUT_; ++j) acc[j] += v * P[n][j];
  }
  #pragma unroll
  for (int j = 0; j < CUT_; ++j) {
    int m  = j * HB_ + h;        // key index 0..2047
    int ch = m >> 5;             // 32-key chunk
    size_t base = ((size_t)b * 64 + ch) * CHUNK_HALFS;
    size_t dst;
    if (!isV) {
      int mtL = (m >> 4) & 1, nn = m & 15;
      int kt = c >> 5, kk = c & 31;
      int lane = nn + 16 * ((kk >> 3) & 1);
      int half = (kk & 7) + 8 * (kk >> 4);
      dst = base + ((size_t)(mtL * 16 + kt) * 32 + lane) * 16 + half;
    } else {
      int kk2 = m & 31;
      int nt = c >> 4, nn2 = c & 15;
      int lane = nn2 + 16 * ((kk2 >> 3) & 1);
      int half = (kk2 & 7) + 8 * (kk2 >> 4);
      dst = base + 16384 + ((size_t)nt * 32 + lane) * 16 + half;
    }
    KVP[dst] = (_Float16)acc[j];
  }
}

// ======================================================================
// 4) flash attention: 4 waves/block, one 16-row Q tile per wave.
//    TDM double-buffers each 64KB K+V chunk into LDS (TENSORcnt),
//    all 4 waves consume it; online softmax via per-wave LDS round-trip.
// ======================================================================
__global__ void __launch_bounds__(128, 1)
attn_kernel(const _Float16* __restrict__ QP, const _Float16* __restrict__ KVP,
            float* __restrict__ out) {
  extern __shared__ _Float16 kvS[];            // 2 x CHUNK_HALFS (128KB)
  __shared__ float    sS[4][16][32];
  __shared__ _Float16 pS[4][16][32];
  __shared__ float    aS[4][16];

  const int w    = threadIdx.x >> 5;
  const int lane = threadIdx.x & 31;
  const int b    = blockIdx.x >> 6;
  const int qt   = ((blockIdx.x & 63) << 2) + w;     // 0..255

  const int g       = lane >> 4;
  const int mm      = lane & 15;
  const int rowHalf = g * 8;

  const _Float16* Qbase = QP + (((size_t)(b * 256 + qt) * 16) * 32 + lane) * 16;
  const _Float16* KVb   = KVP + (size_t)b * 64 * CHUNK_HALFS;
  const unsigned  kvOff = (unsigned)(uintptr_t)(void*)kvS;   // LDS byte address

  const v8f vzero = {0.f, 0.f, 0.f, 0.f, 0.f, 0.f, 0.f, 0.f};
  v8f acc[32];
  #pragma unroll
  for (int t = 0; t < 32; ++t) acc[t] = vzero;

  float m_run = -3.0e38f;
  float l_run = 0.f;
  const float invT = 1.0f / TEMP_;

  // prologue: stream chunk 0 into buffer 0
  if (w == 0) tdm_load_1d(kvOff, KVb, CHUNK_HALFS);

  for (int ch = 0; ch < 64; ++ch) {
    if (w == 0) __builtin_amdgcn_s_wait_tensorcnt(0);  // buf[ch&1] landed
    __syncthreads();                                   // visible to all waves
    if (w == 0 && ch < 63)                             // stream next chunk
      tdm_load_1d(kvOff + (unsigned)(((ch + 1) & 1) * CHUNK_BYTES),
                  KVb + (size_t)(ch + 1) * CHUNK_HALFS, CHUNK_HALFS);

    const _Float16* buf = kvS + (size_t)(ch & 1) * CHUNK_HALFS;

    // ---- S = Q K^T for keys [ch*32, ch*32+32), two 16x16 N-tiles ----
    v8f s0 = vzero, s1 = vzero;
    v16h a_c = ld16(Qbase);                            // pipelined Q A-frags
    #pragma unroll
    for (int kt = 0; kt < 16; ++kt) {
      v16h a_n = (kt < 15) ? ld16(Qbase + (size_t)(kt + 1) * 512) : a_c;
      v16h b0 = ld16(buf + ((size_t)kt * 32 + lane) * 16);
      v16h b1 = ld16(buf + ((size_t)(16 + kt) * 32 + lane) * 16);
      s0 = __builtin_amdgcn_wmma_f32_16x16x32_f16(false, a_c, false, b0, (short)0, s0, false, false);
      s1 = __builtin_amdgcn_wmma_f32_16x16x32_f16(false, a_c, false, b1, (short)0, s1, false, false);
      a_c = a_n;
    }

    // stage S tile to LDS (C-layout -> row major [16][32])
    #pragma unroll
    for (int r = 0; r < 8; ++r) {
      sS[w][r + rowHalf][mm]      = s0[r];
      sS[w][r + rowHalf][mm + 16] = s1[r];
    }

    // ---- online softmax: lanes 0..15 each own one row ----
    if (lane < 16) {
      float mx = m_run;
      #pragma unroll
      for (int x = 0; x < 32; ++x) mx = fmaxf(mx, sS[w][lane][x] * invT);
      float alphaR = __expf(m_run - mx);
      float sum = 0.f;
      #pragma unroll
      for (int x = 0; x < 32; ++x) {
        float p = __expf(sS[w][lane][x] * invT - mx);
        sum += p;
        pS[w][lane][x] = (_Float16)p;
      }
      l_run = l_run * alphaR + sum;
      m_run = mx;
      aS[w][lane] = alphaR;
    }
    // broadcast per-row alpha, rescale accumulators (same-wave LDS is in-order)
    float av[8];
    #pragma unroll
    for (int r = 0; r < 8; ++r) av[r] = aS[w][rowHalf + r];
    #pragma unroll
    for (int t = 0; t < 32; ++t) {
      #pragma unroll
      for (int r = 0; r < 8; ++r) acc[t][r] *= av[r];
    }

    // ---- P * V : A-frag of P straight out of row-major LDS tile ----
    v8h plo = *(const v8h*)&pS[w][mm][8 * g];
    v8h phi = *(const v8h*)&pS[w][mm][16 + 8 * g];
    v16h pa = cat16(plo, phi);
    const _Float16* vb0 = buf + 16384 + (size_t)lane * 16;
    #pragma unroll
    for (int nt = 0; nt < 32; ++nt) {
      v16h bv = ld16(vb0 + (size_t)nt * 512);
      acc[nt] = __builtin_amdgcn_wmma_f32_16x16x32_f16(false, pa, false, bv, (short)0, acc[nt], false, false);
    }
  }

  // ---- epilogue: divide by row sums, store [b][h][c] ----
  if (lane < 16) aS[w][lane] = 1.0f / l_run;
  float av[8];
  #pragma unroll
  for (int r = 0; r < 8; ++r) av[r] = aS[w][rowHalf + r];
  float* ob = out + ((size_t)b * H_ + qt * 16) * C_;
  #pragma unroll
  for (int t = 0; t < 32; ++t) {
    #pragma unroll
    for (int r = 0; r < 8; ++r) {
      ob[(size_t)(r + rowHalf) * C_ + t * 16 + mm] = acc[t][r] * av[r];
    }
  }
}

// ======================================================================
// 5) deterministic instance norm: two-level tree reduction + normalize
// ======================================================================
__global__ void partial_kernel(const float* __restrict__ val, float* __restrict__ part) {
  __shared__ float s1[256], s2[256];
  const int bid = blockIdx.x;                 // B*256
  const int t = threadIdx.x;
  const float* p = val + (size_t)bid * 8192;
  float a = 0.f, b2 = 0.f;
  for (int i = t; i < 8192; i += 256) { float v = p[i]; a += v; b2 += v * v; }
  s1[t] = a; s2[t] = b2; __syncthreads();
  for (int o = 128; o > 0; o >>= 1) {
    if (t < o) { s1[t] += s1[t + o]; s2[t] += s2[t + o]; }
    __syncthreads();
  }
  if (t == 0) { part[bid * 2] = s1[0]; part[bid * 2 + 1] = s2[0]; }
}

__global__ void stats_kernel(const float* __restrict__ part, float* __restrict__ stats) {
  __shared__ float s1[256], s2[256];
  const int b = blockIdx.x, t = threadIdx.x;
  s1[t] = part[(b * 256 + t) * 2];
  s2[t] = part[(b * 256 + t) * 2 + 1];
  __syncthreads();
  for (int o = 128; o > 0; o >>= 1) {
    if (t < o) { s1[t] += s1[t + o]; s2[t] += s2[t + o]; }
    __syncthreads();
  }
  if (t == 0) {
    const float n = (float)((size_t)H_ * C_);
    float mean = s1[0] / n;
    float var  = s2[0] / n - mean * mean;
    stats[b * 2]     = mean;
    stats[b * 2 + 1] = rsqrtf(var + 1e-5f);
  }
}

__global__ void norm_kernel(float* __restrict__ val, const float* __restrict__ stats) {
  size_t idx = (size_t)blockIdx.x * 256 + threadIdx.x;
  int b = (int)(idx >> 21);                  // / (4096*512)
  float mean = stats[b * 2], istd = stats[b * 2 + 1];
  val[idx] = (val[idx] - mean) * istd;
}

// ======================================================================
extern "C" void kernel_launch(void* const* d_in, const int* in_sizes, int n_in,
                              void* d_out, int out_size, void* d_ws, size_t ws_size,
                              hipStream_t stream) {
  const float* q = (const float*)d_in[0];
  const float* k = (const float*)d_in[1];
  const float* v = (const float*)d_in[2];
  char* ws = (char*)d_ws;

  float*    sq    = (float*)(ws + OFF_SQ);
  float*    sk    = (float*)(ws + OFF_SK);
  float*    perm8 = (float*)(ws + OFF_PERM);
  float*    part  = (float*)(ws + OFF_PART);
  float*    stats = (float*)(ws + OFF_STAT);
  _Float16* QP    = (_Float16*)(ws + OFF_QP);
  _Float16* KVP   = (_Float16*)(ws + OFF_KVP);
  float*    out   = (float*)d_out;

  // allow 128KB dynamic LDS for the attention kernel (WGP has 320KB)
  (void)hipFuncSetAttribute((const void*)attn_kernel,
                            hipFuncAttributeMaxDynamicSharedMemorySize, 2 * CHUNK_BYTES);

  mean_kernel    <<<120,   256, 0, stream>>>(q, k, sq, sk);
  sinkhorn_kernel<<<B_,    256, 0, stream>>>(sq, sk, perm8);
  qpack_kernel   <<<65536, 256, 0, stream>>>(q, QP);
  kvpack_kernel  <<<4096,  256, 0, stream>>>(k, perm8, KVP, 0);
  kvpack_kernel  <<<4096,  256, 0, stream>>>(v, perm8, KVP, 1);
  attn_kernel    <<<512,   128, 2 * CHUNK_BYTES, stream>>>(QP, KVP, out);
  partial_kernel <<<2048,  256, 0, stream>>>(out, part);
  stats_kernel   <<<B_,    256, 0, stream>>>(part, stats);
  norm_kernel    <<<65536, 256, 0, stream>>>(out, stats);
}
